// SITH_22041772163434
// MI455X (gfx1250) — matrix operand order repaired and checked by reading, not compiled, not run on hardware
//
#include <hip/hip_runtime.h>

// ---------------- problem constants ----------------
#define NTAU_N 408     // raw n dimension
#define NPAD   416     // padded to 13*32 for K of WMMA
#define OUT_O  100     // outputs
#define OPAD   112     // padded to 7*16 for M of WMMA
#define SEQ_T  512
#define FEAT   512
#define CHUNK  64      // timesteps per chunk
#define NCHUNK 8
#define KT     13      // NPAD/32 k-tiles
#define OT     7       // OPAD/16 o-tiles
#define RPT    26      // n-rows per thread in the scan role (NPAD*16/256)

typedef __attribute__((ext_vector_type(16))) __bf16 v16bf;
typedef __attribute__((ext_vector_type(8)))  float  v8f;

__device__ __forceinline__ unsigned short f2bf(float x) {
  unsigned int u = __float_as_uint(x);
  u += 0x7FFFu + ((u >> 16) & 1u);          // round-to-nearest-even
  return (unsigned short)(u >> 16);
}

// pack two floats into two bf16 in one dword (lo = first arg)
__device__ __forceinline__ unsigned int pack_bf16(float lo, float hi) {
#if __has_builtin(__builtin_amdgcn_cvt_pk_bf16_f32)
  auto p = __builtin_amdgcn_cvt_pk_bf16_f32(lo, hi);
  return __builtin_bit_cast(unsigned int, p);
#else
  return (unsigned int)f2bf(lo) | ((unsigned int)f2bf(hi) << 16);
#endif
}

// ---------------- kernel 1: fold constants, pad, convert weights to bf16 ----
__global__ void sith_prep(const float* __restrict__ invL,
                          const float* __restrict__ subset,
                          const float* __restrict__ d,
                          const float* __restrict__ decay,
                          unsigned short* __restrict__ Wbf,
                          float* __restrict__ dpad,
                          float* __restrict__ cpad) {
  int idx = blockIdx.x * 256 + threadIdx.x;
  const int wtot = OPAD * NPAD;
  if (idx < wtot) {
    int o = idx / NPAD, n = idx % NPAD;
    float w = 0.0f;
    if (o < OUT_O && n < NTAU_N) w = subset[o] * invL[o * NTAU_N + n];
    Wbf[idx] = f2bf(w);
  } else if (idx < wtot + NPAD) {
    int n = idx - wtot;
    const float kAlpha = 1.0f;               // ALPHA
    const float kScale = 0.1f / 0.1f;        // TAU_0 / DT
    dpad[n] = (n < NTAU_N) ? d[n] : 0.0f;
    cpad[n] = (n < NTAU_N) ? (decay[n] * kAlpha * kScale) : 0.0f;
  }
}

// ---------------- kernel 2: per-chunk zero-state local response ------------
__global__ void sith_chunk_local(const float* __restrict__ inp,
                                 const float* __restrict__ d,
                                 const float* __restrict__ decay,
                                 float* __restrict__ E) {
  long idx = (long)blockIdx.x * 256 + threadIdx.x;   // NCHUNK*NPAD*FEAT
  int f = (int)(idx % FEAT);
  int n = (int)((idx / FEAT) % NPAD);
  int j = (int)(idx / ((long)FEAT * NPAD));
  float s = 0.0f;
  if (n < NTAU_N) {
    float dn = d[n], cn = decay[n];
    const float* x = inp + (long)j * CHUNK * FEAT + f;
    for (int i = 0; i < CHUNK; ++i) s = dn * s + cn * x[(long)i * FEAT];
  }
  E[idx] = s;
}

// ---------------- kernel 3: exclusive scan over chunks ---------------------
__global__ void sith_chunk_scan(const float* __restrict__ d,
                                float* __restrict__ E) {
  int idx = blockIdx.x * 256 + threadIdx.x;  // NPAD*FEAT
  int f = idx % FEAT;
  int n = idx / FEAT;
  float d64 = 0.0f;
  if (n < NTAU_N) { float x = d[n]; for (int i = 0; i < 6; ++i) x *= x; d64 = x; }
  float carry = 0.0f;
  for (int j = 0; j < NCHUNK; ++j) {
    long off = ((long)j * NPAD + n) * FEAT + f;
    float tmp = E[off];
    E[off] = carry;
    carry = d64 * carry + tmp;
  }
}

// ---------------- kernel 4: fused scan + bf16-WMMA projection --------------
// grid = (FEAT/16 f-tiles, NCHUNK chunks), block = 256 (8 waves)
__global__ __launch_bounds__(256, 1)
void sith_main(const float* __restrict__ inp,
               const unsigned short* __restrict__ Wbf,
               const float* __restrict__ dpad,
               const float* __restrict__ cpad,
               const float* __restrict__ E,
               float* __restrict__ out) {
  __shared__ unsigned short Sb[2][16 * NPAD];  // double-buffered bf16 state [f][n]
  __shared__ float Xt[CHUNK * 16];             // staged input tile, 4 KB

  const int tid   = threadIdx.x;
  const int lane  = tid & 31;
  const int wave  = tid >> 5;
  const int f0    = blockIdx.x * 16;
  const int chunk = blockIdx.y;
  const int fl    = tid & 15;    // feature column owned in state-update role
  const int nb    = tid >> 4;    // n block (owns n = nb*26 .. nb*26+25)
  const int g     = lane >> 4;   // lane half-group
  const int nl    = lane & 15;

  // --- weight-stationary A fragments (contiguous k = 16*g + e, same order as B) ---
  v16bf A[KT];
  if (wave < OT) {
    int orow = wave * 16 + nl;                     // M = lane&15
    for (int kt = 0; kt < KT; ++kt) {
      union { v16bf v; unsigned short us[16]; } ua;
#pragma unroll
      for (int e = 0; e < 16; ++e) {
        int v4 = e >> 1, p = e & 1;
        int kl = 16 * g + 2 * v4 + p;
        ua.us[e] = Wbf[orow * NPAD + kt * 32 + kl];
      }
      A[kt] = ua.v;
    }
  }

  // --- register-resident recurrence state: contiguous run of 26 n-rows ---
  float s[RPT], dr[RPT], cr[RPT];
#pragma unroll
  for (int r = 0; r < RPT; ++r) {
    int n = nb * RPT + r;
    dr[r] = dpad[n];
    cr[r] = cpad[n];
    s[r]  = E[((long)chunk * NPAD + n) * FEAT + f0 + fl];
  }

  // --- stage this chunk's input tile (64 x 16) into LDS ---
  for (int q = 0; q < 4; ++q) {
    int i = tid + 256 * q;                 // 0..1023
    int t = i >> 4, f = i & 15;
    Xt[i] = inp[((long)(chunk * CHUNK + t)) * FEAT + f0 + f];
  }
  __syncthreads();

  // dword index of this thread's packed-state base within an Sb buffer
  const int wbase = (fl * NPAD + nb * RPT) >> 1;          // RPT even, NPAD even
  // dword index base for this lane's B-fragment reads (per buffer)
  const int rbase = (nl * NPAD + g * 16) >> 1;

  for (int t = 0; t < CHUNK; ++t) {
    // phase 1: advance states, publish packed bf16 copies (13 b32 stores)
    unsigned int* wb = (unsigned int*)Sb[t & 1];
    float x = Xt[t * 16 + fl];
#pragma unroll
    for (int r = 0; r < RPT; r += 2) {
      s[r]     = dr[r]     * s[r]     + cr[r]     * x;
      s[r + 1] = dr[r + 1] * s[r + 1] + cr[r + 1] * x;
      wb[wbase + (r >> 1)] = pack_bf16(s[r], s[r + 1]);
    }
    __syncthreads();   // single barrier per step (readers use the other buffer next)

    // phase 2: projection GEMM, one 16(o)x16(f) tile per wave
    if (wave < OT) {
      const unsigned int* rb = (const unsigned int*)Sb[t & 1] + rbase;
      union Bu { v16bf v; unsigned int ui[8]; } b[2];
      // depth-2 prefetch: tiles 0 and 1 in flight before first WMMA
#pragma unroll
      for (int h = 0; h < 8; ++h) b[0].ui[h] = rb[h];
#pragma unroll
      for (int h = 0; h < 8; ++h) b[1].ui[h] = rb[16 + h];
      v8f acc0 = {}, acc1 = {};
#pragma unroll
      for (int kt = 0; kt < KT; ++kt) {
        int cur = kt & 1;
        if (cur)
          acc1 = __builtin_amdgcn_wmma_f32_16x16x32_bf16(
              false, A[kt], false, b[1].v, (short)0, acc1, false, false);
        else
          acc0 = __builtin_amdgcn_wmma_f32_16x16x32_bf16(
              false, A[kt], false, b[0].v, (short)0, acc0, false, false);
        if (kt + 2 < KT) {
          const unsigned int* nx = rb + (kt + 2) * 16;   // 32 ushorts per k-tile
#pragma unroll
          for (int h = 0; h < 8; ++h) b[cur].ui[h] = nx[h];
        }
      }
      // pin the pipeline: [DS_READ x4][(WMMA x1)(DS_READ x2)]x11 [WMMA x2]
      __builtin_amdgcn_sched_group_barrier(0x100, 4, 0);   // 4 ds_load (tiles 0,1)
#pragma unroll
      for (int i = 0; i < 11; ++i) {
        __builtin_amdgcn_sched_group_barrier(0x008, 1, 0); // 1 wmma
        __builtin_amdgcn_sched_group_barrier(0x100, 2, 0); // 2 ds_load (tile kt+2)
      }
      __builtin_amdgcn_sched_group_barrier(0x008, 2, 0);   // last 2 wmma

      v8f acc = acc0 + acc1;
      int tg = chunk * CHUNK + t;
#pragma unroll
      for (int r = 0; r < 8; ++r) {
        int o = wave * 16 + 8 * g + r;     // C/D layout: M = r + 8*laneGroup
        if (o < OUT_O)
          out[((long)tg * OUT_O + o) * FEAT + f0 + nl] = acc[r];
      }
    }
  }
}

// ---------------- host launcher --------------------------------------------
extern "C" void kernel_launch(void* const* d_in, const int* in_sizes, int n_in,
                              void* d_out, int out_size, void* d_ws, size_t ws_size,
                              hipStream_t stream) {
  const float* inp    = (const float*)d_in[0];
  const float* invL   = (const float*)d_in[1];
  const float* d      = (const float*)d_in[2];
  const float* decay  = (const float*)d_in[3];
  const float* subset = (const float*)d_in[4];
  float* out = (float*)d_out;

  char* ws = (char*)d_ws;
  unsigned short* Wbf = (unsigned short*)ws;            //  93,184 B
  float* dpad = (float*)(ws + 98304);                   //   1,664 B
  float* cpad = (float*)(ws + 98304 + 2048);            //   1,664 B
  float* E    = (float*)(ws + 102400);                  // 6,815,744 B

  {
    int total = OPAD * NPAD + NPAD;
    sith_prep<<<(total + 255) / 256, 256, 0, stream>>>(invL, subset, d, decay,
                                                       Wbf, dpad, cpad);
  }
  {
    long total = (long)NCHUNK * NPAD * FEAT;            // 1,703,936
    sith_chunk_local<<<(int)(total / 256), 256, 0, stream>>>(inp, d, decay, E);
  }
  sith_chunk_scan<<<(NPAD * FEAT) / 256, 256, 0, stream>>>(d, E);

  dim3 grid(FEAT / 16, NCHUNK);
  sith_main<<<grid, 256, 0, stream>>>(inp, Wbf, dpad, cpad, E, out);
}
